// TransformerBlock_20693152432464
// MI455X (gfx1250) — matrix operand order repaired and checked by reading, not compiled, not run on hardware
//
#include <hip/hip_runtime.h>

// ---------------------------------------------------------------------------
// MI455X (gfx1250, wave32) transformer block: bf16 WMMA compute everywhere,
// async global->LDS staging (ASYNCcnt) with double-buffered GEMM tiles.
// ---------------------------------------------------------------------------

typedef unsigned short u16;
typedef __attribute__((ext_vector_type(16))) __bf16       bf16x16;
typedef __attribute__((ext_vector_type(8)))  float        f32x8;
typedef __attribute__((ext_vector_type(4)))  unsigned int u32x4;

union Frag {             // 32 bytes: one WMMA A/B operand (16 bf16 per lane)
    u32x4   q[2];
    bf16x16 v;
};

// --- CDNA5 async global->LDS copy (16B per lane), with sync fallback -------
#ifndef __has_builtin
#define __has_builtin(x) 0
#endif

#if __has_builtin(__builtin_amdgcn_global_load_async_to_lds_b128)
typedef int v4i __attribute__((vector_size(16)));
typedef __attribute__((address_space(1))) v4i v4i_as1;   // global pointee
typedef __attribute__((address_space(3))) v4i v4i_as3;   // LDS pointee
#define COPY16(l, g)                                                          \
    __builtin_amdgcn_global_load_async_to_lds_b128((v4i_as1*)(g),             \
                                                   (v4i_as3*)(l), 0, 0)
#if __has_builtin(__builtin_amdgcn_s_wait_asynccnt)
#define ASYNC_WAIT() __builtin_amdgcn_s_wait_asynccnt(0)
#else
#define ASYNC_WAIT() asm volatile("s_wait_asynccnt 0x0" ::: "memory")
#endif
#else
#define COPY16(l, g) (*(u32x4*)(l) = *(const u32x4*)(g))
#define ASYNC_WAIT() ((void)0)
#endif

__device__ __forceinline__ u16 f2bf(float f) {
    unsigned int u = __builtin_bit_cast(unsigned int, f);
    unsigned int r = u + 0x7FFFu + ((u >> 16) & 1u);   // round-to-nearest-even
    return (u16)(r >> 16);
}

__device__ __forceinline__ f32x8 wmma_bf16(const Frag& a, const Frag& b, f32x8 c) {
    return __builtin_amdgcn_wmma_f32_16x16x32_bf16(
        false, a.v, false, b.v, (short)0, c, false, false);
}

// ---------------------------------------------------------------------------
// Scalars: lambda = clip(exp(l1)-exp(l2)+0.2, 0.1, 5)   (layer_idx = 0)
// ---------------------------------------------------------------------------
__global__ void lambda_kernel(const float* lq1, const float* lk1,
                              const float* lq2, const float* lk2, float* lam) {
    __shared__ float r1[64], r2[64];
    int t = threadIdx.x;
    r1[t] = lq1[t] * lk1[t];
    r2[t] = lq2[t] * lk2[t];
    __syncthreads();
    for (int s = 32; s > 0; s >>= 1) {
        if (t < s) { r1[t] += r1[t + s]; r2[t] += r2[t + s]; }
        __syncthreads();
    }
    if (t == 0) {
        float l1 = fminf(fmaxf(r1[0], -10.0f), 10.0f);
        float l2 = fminf(fmaxf(r2[0], -10.0f), 10.0f);
        float lam_init = 0.8f - 0.6f;                    // 0.8 - 0.6*exp(0)
        *lam = fminf(fmaxf(__expf(l1) - __expf(l2) + lam_init, 0.1f), 5.0f);
    }
}

// ---------------------------------------------------------------------------
// GRN: deterministic two-stage per-batch sum-of-squares, then nx scalar.
// ---------------------------------------------------------------------------
__global__ void __launch_bounds__(256) sumsq_partial(const float* __restrict__ x,
                                                     float* __restrict__ partial) {
    __shared__ float red[256];
    int b  = blockIdx.x >> 10;
    int bi = blockIdx.x & 1023;
    size_t base = ((size_t)b << 20) + ((size_t)bi << 10);   // S*D = 2^20 per batch
    float s = 0.0f;
#pragma unroll
    for (int j = 0; j < 4; ++j) {
        float v = x[base + threadIdx.x + 256 * j];
        s += v * v;
    }
    red[threadIdx.x] = s;
    __syncthreads();
    for (int st = 128; st > 0; st >>= 1) {
        if ((int)threadIdx.x < st) red[threadIdx.x] += red[threadIdx.x + st];
        __syncthreads();
    }
    if (threadIdx.x == 0) partial[blockIdx.x] = red[0];
}

__global__ void grn_finalize(const float* __restrict__ partial, float* __restrict__ nx) {
    __shared__ float red[256];
    int b = blockIdx.x;
    float s = 0.0f;
#pragma unroll
    for (int j = 0; j < 4; ++j) s += partial[b * 1024 + threadIdx.x + 256 * j];
    red[threadIdx.x] = s;
    __syncthreads();
    for (int st = 128; st > 0; st >>= 1) {
        if ((int)threadIdx.x < st) red[threadIdx.x] += red[threadIdx.x + st];
        __syncthreads();
    }
    if (threadIdx.x == 0) {
        float ss = fminf(fmaxf(red[0], 1e-6f), 1000.0f);
        float g  = sqrtf(ss);
        float n  = g / (g + 1e-6f);                // mean over (B,1,1) last dim = identity
        nx[b] = fminf(fmaxf(n, -1000.0f), 1000.0f);
    }
}

// h = gamma[d]*(x*nx[b]) + beta[d] + x ; emit bf16 for the GEMMs
__global__ void __launch_bounds__(256) grn_apply(const float* __restrict__ x,
                                                 const float* __restrict__ nx,
                                                 const float* __restrict__ gamma,
                                                 const float* __restrict__ beta,
                                                 u16* __restrict__ out) {
    size_t i = (size_t)blockIdx.x * 256 + threadIdx.x;
    int d = (int)(i & 1023);
    int b = (int)(i >> 20);
    float v  = x[i];
    float hv = gamma[d] * (v * nx[b]) + beta[d] + v;
    out[i] = f2bf(hv);
}

// fp32 [K][N] weight -> bf16 transposed [N][K] (GEMM B-tile loads == A-tile loads)
__global__ void __launch_bounds__(256) cvt_transpose(const float* __restrict__ in,
                                                     u16* __restrict__ out,
                                                     int K, int N) {
    size_t i = (size_t)blockIdx.x * 256 + threadIdx.x;
    if (i >= (size_t)K * N) return;
    int k = (int)(i / (size_t)N);
    int n = (int)(i % (size_t)N);
    out[(size_t)n * K + k] = f2bf(in[i]);
}

// ---------------------------------------------------------------------------
// GEMM: C[M,N] = A[M,K](bf16) @ Bt[N,K]^T(bf16), fp32 accumulate via WMMA.
// 128x128 tile, 8 waves (4x2 of 32x64), K-steps of 32. Double-buffered LDS
// staged by async global->LDS b128 copies; next tile's DMA overlaps current
// tile's WMMAs. LDS rows padded to 40 bf16 for 16B-aligned fragment reads.
// ---------------------------------------------------------------------------
template <bool HAS_BIAS, bool HAS_RES, bool SILU, bool OUT_BF16>
__global__ void __launch_bounds__(256) gemm_bf16_kernel(
    const u16* __restrict__ A, const u16* __restrict__ Bt,
    const float* __restrict__ bias, const float* __restrict__ res,
    void* __restrict__ outp, int M, int N, int K) {
    __shared__ __align__(16) u16 As[2][128 * 40];
    __shared__ __align__(16) u16 Bs[2][128 * 40];

    const int tid  = threadIdx.x;
    const int lane = tid & 31;
    const int wv   = tid >> 5;
    const int wm   = wv >> 1;        // 0..3 -> 32-row band
    const int wn   = wv & 1;         // 0..1 -> 64-col band
    const int half = lane >> 4;
    const int lm   = lane & 15;
    const int m0 = blockIdx.y * 128;
    const int n0 = blockIdx.x * 128;

    // per-thread 16B chunk mapping: 512 chunks per 128x32 tile, 2 per thread
    const int rr = tid >> 2;               // row 0..63 (and +64)
    const int cc = (tid & 3) << 3;         // elem col 0,8,16,24

    f32x8 acc[2][4];
#pragma unroll
    for (int mi = 0; mi < 2; ++mi)
#pragma unroll
        for (int ni = 0; ni < 4; ++ni)
#pragma unroll
            for (int r = 0; r < 8; ++r) acc[mi][ni][r] = 0.0f;

    auto issue = [&](int buf, int k0) {
#pragma unroll
        for (int i = 0; i < 2; ++i) {
            int r = rr + 64 * i;
            COPY16(&As[buf][r * 40 + cc], A  + (size_t)(m0 + r) * K + k0 + cc);
            COPY16(&Bs[buf][r * 40 + cc], Bt + (size_t)(n0 + r) * K + k0 + cc);
        }
    };

    issue(0, 0);
    ASYNC_WAIT();
    __syncthreads();

    const int nk = K >> 5;
    for (int kt = 0; kt < nk; ++kt) {
        const int cur = kt & 1;
        if (kt + 1 < nk) issue(cur ^ 1, (kt + 1) << 5);   // overlap DMA w/ WMMA

        const u16* Ac = As[cur];
        const u16* Bc = Bs[cur];
        Frag a0, a1;
        {   // A-layout: lanes 0-15 hold K {0..7,16..23}, lanes 16-31 K {8..15,24..31}
            const u16* p0 = Ac + (wm * 32 + lm) * 40 + half * 8;
            a0.q[0] = *(const u32x4*)(p0);
            a0.q[1] = *(const u32x4*)(p0 + 16);
            const u16* p1 = Ac + (wm * 32 + 16 + lm) * 40 + half * 8;
            a1.q[0] = *(const u32x4*)(p1);
            a1.q[1] = *(const u32x4*)(p1 + 16);
        }
#pragma unroll
        for (int ni = 0; ni < 4; ++ni) {
            Frag bfrag;  // B-layout: lanes 0-15 N=lm K {0..7,16..23}, lanes 16-31 N=lm K {8..15,24..31}
            const u16* bp = Bc + (wn * 64 + ni * 16 + lm) * 40 + half * 8;
            bfrag.q[0] = *(const u32x4*)(bp);
            bfrag.q[1] = *(const u32x4*)(bp + 16);
            acc[0][ni] = wmma_bf16(a0, bfrag, acc[0][ni]);
            acc[1][ni] = wmma_bf16(a1, bfrag, acc[1][ni]);
        }
        ASYNC_WAIT();
        __syncthreads();
    }

    float* outf = (float*)outp;
    u16*   outb = (u16*)outp;
#pragma unroll
    for (int mi = 0; mi < 2; ++mi)
#pragma unroll
        for (int ni = 0; ni < 4; ++ni)
#pragma unroll
            for (int r = 0; r < 8; ++r) {
                int row = m0 + wm * 32 + mi * 16 + r + 8 * half;
                int col = n0 + wn * 64 + ni * 16 + lm;
                float v = acc[mi][ni][r];
                if (HAS_BIAS) v += bias[col];
                if (SILU)     v = v * (1.0f / (1.0f + __expf(-v)));
                if (HAS_RES)  v += res[(size_t)row * N + col];
                size_t o = (size_t)row * N + col;
                if (OUT_BF16) outb[o] = f2bf(v);
                else          outf[o] = v;
            }
}

// ---------------------------------------------------------------------------
// Differential flash attention. One workgroup = (b, h, 64-query tile),
// 4 waves x 16 query rows. 64-key K1/K2 tiles staged by async global->LDS,
// V transposed manually. Online clipped softmax with shfl row reductions,
// P bounced through LDS (C->A layout), out = o1/l1 - lam*o2/l2 -> bf16.
// ---------------------------------------------------------------------------
__device__ __forceinline__ void online_softmax(f32x8 (&s)[4], float (&m)[8],
                                               float (&l)[8], f32x8 (&o)[4],
                                               u16* pstage, int half, int lm) {
    const float scale = 0.125f;                       // hd^-0.5
#pragma unroll
    for (int r = 0; r < 8; ++r) {
        float tmax = -1e30f;
#pragma unroll
        for (int n = 0; n < 4; ++n) {
            float sv = s[n][r] * scale;
            sv = fminf(fmaxf(sv, -100.0f), 100.0f);   // ATTN_CLIP
            s[n][r] = sv;
            tmax = fmaxf(tmax, sv);
        }
        tmax = fmaxf(tmax, __shfl_xor(tmax, 1, 32));
        tmax = fmaxf(tmax, __shfl_xor(tmax, 2, 32));
        tmax = fmaxf(tmax, __shfl_xor(tmax, 4, 32));
        tmax = fmaxf(tmax, __shfl_xor(tmax, 8, 32));
        float mn   = fmaxf(m[r], tmax);
        float corr = __expf(m[r] - mn);
        float rs   = 0.0f;
#pragma unroll
        for (int n = 0; n < 4; ++n) {
            float p = __expf(s[n][r] - mn);
            rs += p;
            pstage[(r + 8 * half) * 72 + n * 16 + lm] = f2bf(p);
        }
        rs += __shfl_xor(rs, 1, 32);
        rs += __shfl_xor(rs, 2, 32);
        rs += __shfl_xor(rs, 4, 32);
        rs += __shfl_xor(rs, 8, 32);
        l[r] = l[r] * corr + rs;
        m[r] = mn;
#pragma unroll
        for (int i = 0; i < 4; ++i) o[i][r] *= corr;
    }
}

__device__ __forceinline__ void pv_accum(f32x8 (&o)[4], const u16* pstage,
                                         const u16* Vts, int half, int lm) {
#pragma unroll
    for (int kk = 0; kk < 2; ++kk) {
        Frag pa;                                    // A-layout read of 16x64 P
        const u16* pp = pstage + lm * 72 + kk * 32 + half * 8;
        pa.q[0] = *(const u32x4*)(pp);
        pa.q[1] = *(const u32x4*)(pp + 16);
#pragma unroll
        for (int i = 0; i < 4; ++i) {
            Frag vb;                                // B-layout: N=dim, K=key
            const u16* vp = Vts + (i * 16 + lm) * 72 + kk * 32 + half * 8;
            vb.q[0] = *(const u32x4*)(vp);
            vb.q[1] = *(const u32x4*)(vp + 16);
            o[i] = wmma_bf16(pa, vb, o[i]);
        }
    }
}

__global__ void __launch_bounds__(128) attn_kernel(const u16* __restrict__ qkv,
                                                   const float* __restrict__ lamp,
                                                   u16* __restrict__ obf) {
    __shared__ __align__(16) u16 K1s[64 * 72];
    __shared__ __align__(16) u16 K2s[64 * 72];
    __shared__ __align__(16) u16 Vts[64 * 72];      // transposed: [dim][key]
    __shared__ __align__(16) u16 Ps[4][16 * 72];    // per-wave P bounce

    const int tid  = threadIdx.x;
    const int lane = tid & 31;
    const int wv   = tid >> 5;
    const int half = lane >> 4;
    const int lm   = lane & 15;
    const int wg = blockIdx.x;
    const int qt = wg & 15;         // 16 query tiles of 64
    const int bh = wg >> 4;
    const int h  = bh & 15;
    const int b  = bh >> 4;

    // Q fragments for this wave's 16 rows (q1: c=0, q2: c=1). Token stride 5120.
    const int    qrow = qt * 64 + wv * 16 + lm;
    const size_t tok  = ((size_t)(b * 1024 + qrow)) * 5120 + (size_t)h * 64;
    Frag q1f[2], q2f[2];
#pragma unroll
    for (int kk = 0; kk < 2; ++kk) {
        int d0 = kk * 32 + half * 8;
        q1f[kk].q[0] = *(const u32x4*)(qkv + tok + d0);
        q1f[kk].q[1] = *(const u32x4*)(qkv + tok + d0 + 16);
        q2f[kk].q[0] = *(const u32x4*)(qkv + tok + 1024 + d0);
        q2f[kk].q[1] = *(const u32x4*)(qkv + tok + 1024 + d0 + 16);
    }

    f32x8 o1[4], o2[4];
    float m1[8], l1[8], m2[8], l2[8];
#pragma unroll
    for (int i = 0; i < 4; ++i)
#pragma unroll
        for (int r = 0; r < 8; ++r) { o1[i][r] = 0.0f; o2[i][r] = 0.0f; }
#pragma unroll
    for (int r = 0; r < 8; ++r) { m1[r] = -1e30f; l1[r] = 0.0f; m2[r] = -1e30f; l2[r] = 0.0f; }

    for (int kt = 0; kt < 16; ++kt) {
        // --- stage K1/K2 via async global->LDS (16B chunks, 4 per thread each)
#pragma unroll
        for (int i = 0; i < 4; ++i) {
            int e  = tid + 128 * i;              // 512 chunks per 64x64 tile
            int ky = e >> 3;
            int c  = (e & 7) << 3;
            size_t src = ((size_t)(b * 1024 + kt * 64 + ky)) * 5120 + (size_t)h * 64 + c;
            COPY16(&K1s[ky * 72 + c], qkv + src + 2048);   // c=2: k1
            COPY16(&K2s[ky * 72 + c], qkv + src + 3072);   // c=3: k2
        }
        // --- stage V transposed (manual; needs element transpose)
#pragma unroll
        for (int i = 0; i < 16; ++i) {
            int e  = tid + 128 * i;              // 2048 u32 per tile
            int ky = e >> 5;
            int d2 = (e & 31) << 1;
            size_t src = ((size_t)(b * 1024 + kt * 64 + ky)) * 5120 + (size_t)h * 64 + d2;
            unsigned int vv = *(const unsigned int*)(qkv + src + 4096);   // c=4: v
            Vts[(d2    ) * 72 + ky] = (u16)(vv & 0xffffu);
            Vts[(d2 + 1) * 72 + ky] = (u16)(vv >> 16);
        }
        ASYNC_WAIT();
        __syncthreads();

        {   // attention branch 1: S1 = Q1 @ K1^T over this key tile
            f32x8 s[4];
#pragma unroll
            for (int n = 0; n < 4; ++n) {
                f32x8 a;
#pragma unroll
                for (int r = 0; r < 8; ++r) a[r] = 0.0f;
#pragma unroll
                for (int kk = 0; kk < 2; ++kk) {
                    Frag kb;
                    const u16* kp = K1s + (n * 16 + lm) * 72 + kk * 32 + half * 8;
                    kb.q[0] = *(const u32x4*)(kp);
                    kb.q[1] = *(const u32x4*)(kp + 16);
                    a = wmma_bf16(q1f[kk], kb, a);
                }
                s[n] = a;
            }
            online_softmax(s, m1, l1, o1, Ps[wv], half, lm);
        }
        __syncthreads();
        pv_accum(o1, Ps[wv], Vts, half, lm);
        __syncthreads();

        {   // attention branch 2
            f32x8 s[4];
#pragma unroll
            for (int n = 0; n < 4; ++n) {
                f32x8 a;
#pragma unroll
                for (int r = 0; r < 8; ++r) a[r] = 0.0f;
#pragma unroll
                for (int kk = 0; kk < 2; ++kk) {
                    Frag kb;
                    const u16* kp = K2s + (n * 16 + lm) * 72 + kk * 32 + half * 8;
                    kb.q[0] = *(const u32x4*)(kp);
                    kb.q[1] = *(const u32x4*)(kp + 16);
                    a = wmma_bf16(q2f[kk], kb, a);
                }
                s[n] = a;
            }
            online_softmax(s, m2, l2, o2, Ps[wv], half, lm);
        }
        __syncthreads();
        pv_accum(o2, Ps[wv], Vts, half, lm);
        __syncthreads();
    }

    const float lam = lamp[0];
#pragma unroll
    for (int i = 0; i < 4; ++i)
#pragma unroll
        for (int r = 0; r < 8; ++r) {
            float v = o1[i][r] / (l1[r] + 1e-6f) - lam * (o2[i][r] / (l2[r] + 1e-6f));
            int rq  = qt * 64 + wv * 16 + r + 8 * half;
            int col = h * 64 + i * 16 + lm;
            obf[((size_t)(b * 1024 + rq)) * 1024 + col] = f2bf(v);
        }
}

// ---------------------------------------------------------------------------
// Host orchestration
// ---------------------------------------------------------------------------
extern "C" void kernel_launch(void* const* d_in, const int* in_sizes, int n_in,
                              void* d_out, int out_size, void* d_ws, size_t ws_size,
                              hipStream_t stream) {
    (void)in_sizes; (void)n_in; (void)out_size; (void)ws_size;

    const float* x      = (const float*)d_in[0];
    const float* w_qkv  = (const float*)d_in[1];
    const float* lq1    = (const float*)d_in[2];
    const float* lk1    = (const float*)d_in[3];
    const float* lq2    = (const float*)d_in[4];
    const float* lk2    = (const float*)d_in[5];
    const float* w_proj = (const float*)d_in[6];
    const float* b_proj = (const float*)d_in[7];
    const float* gamma1 = (const float*)d_in[8];
    const float* beta1  = (const float*)d_in[9];
    const float* gamma2 = (const float*)d_in[10];
    const float* beta2  = (const float*)d_in[11];
    const float* w1     = (const float*)d_in[12];
    const float* b1     = (const float*)d_in[13];
    const float* w2     = (const float*)d_in[14];
    const float* b2     = (const float*)d_in[15];
    float* out = (float*)d_out;

    // workspace carve-up (256B aligned); total ~132 MB
    char*  w   = (char*)d_ws;
    size_t off = 0;
    auto carve = [&](size_t bytes) -> void* {
        void* p = w + off;
        off += (bytes + 255) & ~(size_t)255;
        return p;
    };
    float* partial = (float*)carve(4096 * 4);                 // sumsq partials
    float* nx      = (float*)carve(4 * 4);
    float* lamw    = (float*)carve(4);
    u16*   hbuf    = (u16*)carve((size_t)4096 * 1024 * 2);    // h1 then h2 (bf16)
    u16*   qkvb    = (u16*)carve((size_t)4096 * 5120 * 2);
    u16*   obf     = (u16*)carve((size_t)4096 * 1024 * 2);
    float* x1      = (float*)carve((size_t)4096 * 1024 * 4);
    u16*   midb    = (u16*)carve((size_t)4096 * 4096 * 2);
    u16*   wqkvb   = (u16*)carve((size_t)5120 * 1024 * 2);    // [N][K] bf16
    u16*   wprojb  = (u16*)carve((size_t)1024 * 1024 * 2);
    u16*   w1b     = (u16*)carve((size_t)4096 * 1024 * 2);
    u16*   w2b     = (u16*)carve((size_t)1024 * 4096 * 2);

    // scalars + weight conversion (transposed bf16)
    lambda_kernel<<<1, 64, 0, stream>>>(lq1, lk1, lq2, lk2, lamw);
    cvt_transpose<<<20480, 256, 0, stream>>>(w_qkv, wqkvb, 1024, 5120);
    cvt_transpose<<<4096,  256, 0, stream>>>(w_proj, wprojb, 1024, 1024);
    cvt_transpose<<<16384, 256, 0, stream>>>(w1, w1b, 1024, 4096);
    cvt_transpose<<<16384, 256, 0, stream>>>(w2, w2b, 4096, 1024);

    // GRN 1 -> h1 (bf16)
    sumsq_partial<<<4096, 256, 0, stream>>>(x, partial);
    grn_finalize<<<4, 256, 0, stream>>>(partial, nx);
    grn_apply<<<16384, 256, 0, stream>>>(x, nx, gamma1, beta1, hbuf);

    // QKV = h1 @ w_qkv  (M=4096, N=5120, K=1024), bf16 out
    gemm_bf16_kernel<false, false, false, true>
        <<<dim3(40, 32), 256, 0, stream>>>(hbuf, wqkvb, nullptr, nullptr,
                                           (void*)qkvb, 4096, 5120, 1024);

    // differential attention -> obf (bf16)
    attn_kernel<<<1024, 128, 0, stream>>>(qkvb, lamw, obf);

    // x1 = obf @ w_proj + b_proj + x  (fp32)
    gemm_bf16_kernel<true, true, false, false>
        <<<dim3(8, 32), 256, 0, stream>>>(obf, wprojb, b_proj, x,
                                          (void*)x1, 4096, 1024, 1024);

    // GRN 2 -> h2 (bf16, reuses hbuf)
    sumsq_partial<<<4096, 256, 0, stream>>>(x1, partial);
    grn_finalize<<<4, 256, 0, stream>>>(partial, nx);
    grn_apply<<<16384, 256, 0, stream>>>(x1, nx, gamma2, beta2, hbuf);

    // mid = silu(h2 @ w1 + b1)  (bf16)
    gemm_bf16_kernel<true, false, true, true>
        <<<dim3(32, 32), 256, 0, stream>>>(hbuf, w1b, b1, nullptr,
                                           (void*)midb, 4096, 4096, 1024);

    // out = mid @ w2 + b2 + x1  (fp32)
    gemm_bf16_kernel<true, true, false, false>
        <<<dim3(8, 32), 256, 0, stream>>>(midb, w2b, b2, x1,
                                          (void*)out, 4096, 1024, 4096);
}
